// NormalizingFlow_40785009443552
// MI455X (gfx1250) — compile-verified, matrix operand order checked by reading, not run on hardware
//
#include <hip/hip_runtime.h>
#include <cstdint>
#include <cstddef>

typedef __bf16 bf16;
typedef __attribute__((ext_vector_type(16))) __bf16 v16bf;
typedef __attribute__((ext_vector_type(8)))  float  v8f;

// Types for the gfx1250 async global->LDS builtin:
// param type per hipcc diagnostic: 'int __vector_size__(4*sizeof(int)) __device__ *'
typedef int v4i_ __attribute__((vector_size(16)));
typedef __attribute__((address_space(1))) v4i_* as1_v4i;
typedef __attribute__((address_space(3))) v4i_* as3_v4i;

#define BB   32
#define CCH  512
#define LLN  336
#define HHD  1024
#define PPL  192
#define NFL  4
#define TTOT 1216   /* H + P */
#define TCV  1215   /* conv output length */

#define CDIV(a,b) (((a)+(b)-1)/(b))

// Async global->LDS staging (gfx1250). Guarded so the file compiles with a
// synchronous fallback if the toolchain lacks the builtins.
#if defined(__HIP_DEVICE_COMPILE__) && defined(__gfx1250__) && \
    __has_builtin(__builtin_amdgcn_global_load_async_to_lds_b128) && \
    __has_builtin(__builtin_amdgcn_s_wait_asynccnt)
#define USE_ASYNC 1
#else
#define USE_ASYNC 0
#endif

// ---------------------------------------------------------------------------
// Elementwise / layout kernels
// ---------------------------------------------------------------------------
__global__ void k_cvt_bf16(const float* __restrict__ in, bf16* __restrict__ out, long n) {
  long i = (long)blockIdx.x * blockDim.x + threadIdx.x;
  if (i < n) out[i] = (bf16)in[i];
}

// x_enc (B,C,L) f32 -> xt (B,L,C) bf16
__global__ void k_transpose_x(const float* __restrict__ x, bf16* __restrict__ xt) {
  long i = (long)blockIdx.x * blockDim.x + threadIdx.x;
  long n = (long)BB * LLN * CCH;
  if (i >= n) return;
  int c = (int)(i % CCH); long r = i / CCH;
  int l = (int)(r % LLN); int b = (int)(r / LLN);
  xt[i] = (bf16)x[((long)(b * CCH + c)) * LLN + l];
}

// conv_w (NF,C,C,2) f32 -> packed (NF, C, 2, C) bf16 : out[((l*C+o)*2+tap)*C + i]
__global__ void k_pack_convw(const float* __restrict__ w, bf16* __restrict__ out) {
  long i = (long)blockIdx.x * blockDim.x + threadIdx.x;
  long n = (long)NFL * CCH * 2 * CCH;
  if (i >= n) return;
  int ii = (int)(i % CCH); long r = i / CCH;
  int tap = (int)(r % 2); r /= 2;
  int o = (int)(r % CCH); int l = (int)(r / CCH);
  out[i] = (bf16)w[((((long)l * CCH + o) * CCH + ii) * 2) + tap];
}

__global__ void k_zero_h(float* __restrict__ h, bf16* __restrict__ hb) {
  int i = blockIdx.x * blockDim.x + threadIdx.x;
  if (i < BB * HHD) { h[i] = 0.f; hb[i] = (bf16)0.f; }
}

// ---------------------------------------------------------------------------
// NT GEMM, bf16 WMMA:  C[zc+m, n] = sum_k A[(za+m)*lda + k] * B[n*ldb + k] + bias[n]
//   za = blockIdx.z * zPadA (rows), zc = blockIdx.z * zStrideC (rows)
// Block: 256 threads = 8 waves; tile BM=128 x BN=64 x BK=64.
// Wave w: wm = w>>1 (4 x 32 rows), wn = w&1 (2 x 32 cols) -> 32x32 per wave,
// 4 accumulators, 8 v_wmma per K-stage, each fragment reused twice.
// ---------------------------------------------------------------------------
#define BM 128
#define BN 64
#define BK 64
#define LDT 72   /* bf16 elements per LDS row (64 + 8 pad, 16B-aligned) */

__device__ __forceinline__ v16bf frag_ld(const bf16* base, int lane, int kk) {
  // ISA wave32 16-bit matrix layout: lanes 0-15 hold K={0..7,16..23}(+kk),
  // lanes 16-31 hold K={8..15,24..31}(+kk); contiguous 8-elem halves -> ds_load_b128.
  int r  = lane & 15;
  int kh = (((lane >> 4) & 1) << 3) + kk;
  const bf16* p = base + r * LDT + kh;
  v16bf v;
#pragma unroll
  for (int j = 0; j < 8; ++j) v[j]     = p[j];
#pragma unroll
  for (int j = 0; j < 8; ++j) v[8 + j] = p[16 + j];
  return v;
}

__global__ __launch_bounds__(256)
void k_gemm_nt(const bf16* __restrict__ A, int lda, int zPadA,
               const bf16* __restrict__ Bm, int ldb,
               const float* __restrict__ bias,
               float* __restrict__ Cc, int ldc, int zStrideC,
               int Mz, int N, int K)   // K % 64 == 0 for all call sites
{
  __shared__ bf16 As[BM][LDT];
  __shared__ bf16 Bs[BN][LDT];
  int tid  = threadIdx.x;
  int lane = tid & 31, w = tid >> 5;
  int wm = w >> 1, wn = w & 1;
  int m0 = blockIdx.y * BM, n0 = blockIdx.x * BN;
  size_t za = (size_t)blockIdx.z * (size_t)zPadA;
  size_t zc = (size_t)blockIdx.z * (size_t)zStrideC;
  const bool full = (m0 + BM <= Mz) && (n0 + BN <= N);

  v8f acc00 = {0.f,0.f,0.f,0.f,0.f,0.f,0.f,0.f};
  v8f acc01 = acc00, acc10 = acc00, acc11 = acc00;

  for (int k0 = 0; k0 < K; k0 += BK) {
    if (full) {
#if USE_ASYNC
      // async DMA staging via GLOBAL_LOAD_ASYNC_TO_LDS_B128, then wait ASYNCcnt
#pragma unroll
      for (int s = 0; s < 4; ++s) {          // A: 128x64 -> 1024 b128 slots
        int slot = tid + s * 256;
        int r = slot >> 3, c8 = (slot & 7) << 3;
        const bf16* gA = A + (za + (size_t)(m0 + r)) * (size_t)lda + k0 + c8;
        __builtin_amdgcn_global_load_async_to_lds_b128(
            (as1_v4i)(void*)gA, (as3_v4i)(void*)&As[r][c8], 0, 0);
      }
#pragma unroll
      for (int s = 0; s < 2; ++s) {          // B: 64x64 -> 512 b128 slots
        int slot = tid + s * 256;
        int r = slot >> 3, c8 = (slot & 7) << 3;
        const bf16* gB = Bm + (size_t)(n0 + r) * (size_t)ldb + k0 + c8;
        __builtin_amdgcn_global_load_async_to_lds_b128(
            (as1_v4i)(void*)gB, (as3_v4i)(void*)&Bs[r][c8], 0, 0);
      }
      __builtin_amdgcn_s_wait_asynccnt(0);
#else
#pragma unroll
      for (int s = 0; s < 4; ++s) {
        int slot = tid + s * 256;
        int r = slot >> 3, c8 = (slot & 7) << 3;
        uint4 va = *(const uint4*)(A + (za + (size_t)(m0 + r)) * (size_t)lda + k0 + c8);
        *(uint4*)(&As[r][c8]) = va;
      }
#pragma unroll
      for (int s = 0; s < 2; ++s) {
        int slot = tid + s * 256;
        int r = slot >> 3, c8 = (slot & 7) << 3;
        uint4 vb = *(const uint4*)(Bm + (size_t)(n0 + r) * (size_t)ldb + k0 + c8);
        *(uint4*)(&Bs[r][c8]) = vb;
      }
#endif
      if (k0 + BK < K && tid < BN) {
        __builtin_prefetch(Bm + (size_t)(n0 + tid) * (size_t)ldb + k0 + BK, 0, 0);
      }
    } else {
      // guarded staging (partial tiles): zero-fill out-of-range rows
#pragma unroll
      for (int s = 0; s < 4; ++s) {
        int slot = tid + s * 256;
        int r = slot >> 3, c8 = (slot & 7) << 3;
        uint4 va = {0u,0u,0u,0u};
        if (m0 + r < Mz)
          va = *(const uint4*)(A + (za + (size_t)(m0 + r)) * (size_t)lda + k0 + c8);
        *(uint4*)(&As[r][c8]) = va;
      }
#pragma unroll
      for (int s = 0; s < 2; ++s) {
        int slot = tid + s * 256;
        int r = slot >> 3, c8 = (slot & 7) << 3;
        uint4 vb = {0u,0u,0u,0u};
        if (n0 + r < N)
          vb = *(const uint4*)(Bm + (size_t)(n0 + r) * (size_t)ldb + k0 + c8);
        *(uint4*)(&Bs[r][c8]) = vb;
      }
    }
    __syncthreads();

#pragma unroll
    for (int kk = 0; kk < BK; kk += 32) {
      v16bf a0 = frag_ld(&As[wm * 32][0],      lane, kk);
      v16bf a1 = frag_ld(&As[wm * 32 + 16][0], lane, kk);
      v16bf b0 = frag_ld(&Bs[wn * 32][0],      lane, kk);
      v16bf b1 = frag_ld(&Bs[wn * 32 + 16][0], lane, kk);
      acc00 = __builtin_amdgcn_wmma_f32_16x16x32_bf16(false, a0, false, b0,
                                                      (short)0, acc00, false, false);
      acc01 = __builtin_amdgcn_wmma_f32_16x16x32_bf16(false, a0, false, b1,
                                                      (short)0, acc01, false, false);
      acc10 = __builtin_amdgcn_wmma_f32_16x16x32_bf16(false, a1, false, b0,
                                                      (short)0, acc10, false, false);
      acc11 = __builtin_amdgcn_wmma_f32_16x16x32_bf16(false, a1, false, b1,
                                                      (short)0, acc11, false, false);
    }
    __syncthreads();
  }

  // C/D layout: lane col = lane&15; rows = vgpr + (lane>=16 ? 8 : 0)
  int lane15 = lane & 15;
  int lhalf  = (lane >> 4) << 3;
  v8f accs[2][2] = {{acc00, acc01}, {acc10, acc11}};
#pragma unroll
  for (int i = 0; i < 2; ++i) {
#pragma unroll
    for (int j = 0; j < 2; ++j) {
      int nn = n0 + wn * 32 + j * 16 + lane15;
      int mb = m0 + wm * 32 + i * 16 + lhalf;
      if (nn < N) {
        float bv = bias ? bias[nn] : 0.f;
#pragma unroll
        for (int r = 0; r < 8; ++r) {
          int m = mb + r;
          if (m < Mz) Cc[(zc + (size_t)m) * (size_t)ldc + nn] = accs[i][j][r] + bv;
        }
      }
    }
  }
}

// ---------------------------------------------------------------------------
// GRU gate update at timestep t (torch gate order r,z,n)
// ---------------------------------------------------------------------------
__global__ void k_gru_gate(const float* __restrict__ xw, const float* __restrict__ gh,
                           float* __restrict__ h, bf16* __restrict__ hb, int t)
{
  int idx = blockIdx.x * blockDim.x + threadIdx.x;   // over B*H
  if (idx >= BB * HHD) return;
  int b = idx >> 10; int j = idx & (HHD - 1);
  const float* xr = xw + ((size_t)(b * LLN + t)) * (3 * HHD);
  const float* gr = gh + (size_t)b * (3 * HHD);
  float ir = xr[j], iz = xr[HHD + j], in_ = xr[2 * HHD + j];
  float hr = gr[j], hz = gr[HHD + j], hn  = gr[2 * HHD + j];
  float r = 1.f / (1.f + __expf(-(ir + hr)));
  float z = 1.f / (1.f + __expf(-(iz + hz)));
  float nv = tanhf(in_ + r * hn);
  float hnew = (1.f - z) * nv + z * h[idx];
  h[idx]  = hnew;
  hb[idx] = (bf16)hnew;
}

// enc[b,c,t] = hT[b,t]*wfx[c] + bfx[c]  -> hflow cols [0,1024)
__global__ void k_enc_build(const float* __restrict__ hT, const float* __restrict__ wfx,
                            const float* __restrict__ bfx,
                            float* __restrict__ hflow, bf16* __restrict__ hflowb)
{
  long i = (long)blockIdx.x * blockDim.x + threadIdx.x;
  long n = (long)BB * CCH * HHD;
  if (i >= n) return;
  int t = (int)(i % HHD); long r = i / HHD;
  int c = (int)(r % CCH); int b = (int)(r / CCH);
  float v = hT[b * HHD + t] * wfx[c] + bfx[c];
  long o = ((long)(b * CCH + c)) * TTOT + t;
  hflow[o] = v; hflowb[o] = (bf16)v;
}

// sample = mu + exp(0.5*softplus(ps))*eps ; write hflow cols [1024,1216)
__global__ void k_sample_init(const float* __restrict__ mu, const float* __restrict__ ps,
                              const float* __restrict__ eps, float* __restrict__ sample,
                              float* __restrict__ hflow, bf16* __restrict__ hflowb)
{
  long i = (long)blockIdx.x * blockDim.x + threadIdx.x;
  long n = (long)BB * CCH * PPL;
  if (i >= n) return;
  float x = ps[i];
  float sig = (x > 20.f) ? x : log1pf(__expf(x));
  float s = mu[i] + __expf(0.5f * sig) * eps[i];
  sample[i] = s;
  long bc = i / PPL; int p = (int)(i % PPL);
  long o = bc * TTOT + HHD + p;
  hflow[o] = s; hflowb[o] = (bf16)s;
}

// hflow (B,C,TTOT) f32 -> ht (B,TTOT,C) bf16
__global__ void k_t1(const float* __restrict__ h, bf16* __restrict__ ht) {
  long i = (long)blockIdx.x * blockDim.x + threadIdx.x;
  long n = (long)BB * TTOT * CCH;
  if (i >= n) return;
  int c = (int)(i % CCH); long r = i / CCH;
  int t = (int)(r % TTOT); int b = (int)(r / TTOT);
  ht[i] = (bf16)h[((long)(b * CCH + c)) * TTOT + t];
}

// hct (B,TCV,C) f32 -> relu + re-pad -> hc (B,C,TTOT) f32 + bf16
__global__ void k_t2(const float* __restrict__ hct, float* __restrict__ hc,
                     bf16* __restrict__ hcb)
{
  long i = (long)blockIdx.x * blockDim.x + threadIdx.x;
  long n = (long)BB * CCH * TTOT;
  if (i >= n) return;
  int tt = (int)(i % TTOT); long r = i / TTOT;
  int o = (int)(r % CCH); int b = (int)(r / CCH);
  int st = tt > 0 ? tt - 1 : 0;
  float v = hct[((long)(b * TCV + st)) * CCH + o];
  v = fmaxf(v, 0.f);
  hc[i] = v; hcb[i] = (bf16)v;
}

// sample' = mu + exp(0.5*sg)*sample*0.1 ; h += 0.1*concat(hc[:,:,:1024], sample')
__global__ void k_flow_pw(const float* __restrict__ hc, const float* __restrict__ mu,
                          const float* __restrict__ sg, float* __restrict__ sample,
                          float* __restrict__ hflow)
{
  long i = (long)blockIdx.x * blockDim.x + threadIdx.x;
  long n = (long)BB * CCH * TTOT;
  if (i >= n) return;
  int tt = (int)(i % TTOT); long bc = i / TTOT;
  if (tt < HHD) {
    hflow[i] += 0.1f * hc[i];
  } else {
    int p = tt - HHD;
    long si = bc * PPL + p;
    float s = mu[si] + __expf(0.5f * sg[si]) * sample[si] * 0.1f;
    sample[si] = s;
    hflow[i] += 0.1f * s;
  }
}

// sample (B,C,P) -> out (B,P,C)
__global__ void k_out_tr(const float* __restrict__ sample, float* __restrict__ out) {
  long i = (long)blockIdx.x * blockDim.x + threadIdx.x;
  long n = (long)BB * PPL * CCH;
  if (i >= n) return;
  int c = (int)(i % CCH); long r = i / CCH;
  int p = (int)(r % PPL); int b = (int)(r / PPL);
  out[i] = sample[((long)(b * CCH + c)) * PPL + p];
}

// ---------------------------------------------------------------------------
// Host orchestration
// ---------------------------------------------------------------------------
extern "C" void kernel_launch(void* const* d_in, const int* in_sizes, int n_in,
                              void* d_out, int out_size, void* d_ws, size_t ws_size,
                              hipStream_t stream)
{
  (void)in_sizes; (void)n_in; (void)out_size; (void)ws_size;
  const float* x_enc   = (const float*)d_in[0];
  const float* eps     = (const float*)d_in[1];
  const float* wih     = (const float*)d_in[2];
  const float* whh     = (const float*)d_in[3];
  const float* bih     = (const float*)d_in[4];
  const float* bhh     = (const float*)d_in[5];
  const float* wfx     = (const float*)d_in[6];
  const float* bfx     = (const float*)d_in[7];
  const float* encmu_w = (const float*)d_in[8];
  const float* encmu_b = (const float*)d_in[9];
  const float* encps_w = (const float*)d_in[10];
  const float* encps_b = (const float*)d_in[11];
  const float* conv_w  = (const float*)d_in[12];
  const float* conv_b  = (const float*)d_in[13];
  const float* mu_w    = (const float*)d_in[14];
  const float* mu_b    = (const float*)d_in[15];
  const float* sg_w    = (const float*)d_in[16];
  const float* sg_b    = (const float*)d_in[17];
  float* out = (float*)d_out;

  char* ws = (char*)d_ws;
  size_t off = 0;
  auto alloc = [&](size_t bytes) -> void* {
    void* p = ws + off;
    off += (bytes + 255) & ~(size_t)255;
    return p;
  };

  bf16*  xt_b     = (bf16*) alloc((size_t)BB*LLN*CCH * 2);
  bf16*  wih_b    = (bf16*) alloc((size_t)3*HHD*CCH  * 2);
  bf16*  whh_b    = (bf16*) alloc((size_t)3*HHD*HHD  * 2);
  bf16*  encmuw_b = (bf16*) alloc((size_t)PPL*HHD    * 2);
  bf16*  encpsw_b = (bf16*) alloc((size_t)PPL*HHD    * 2);
  bf16*  muw_b    = (bf16*) alloc((size_t)NFL*PPL*TTOT * 2);
  bf16*  sgw_b    = (bf16*) alloc((size_t)NFL*PPL*TTOT * 2);
  bf16*  convp_b  = (bf16*) alloc((size_t)NFL*CCH*2*CCH * 2);
  float* xw       = (float*)alloc((size_t)BB*LLN*3*HHD * 4);
  float* hgru     = (float*)alloc((size_t)BB*HHD * 4);
  bf16*  hgru_b   = (bf16*) alloc((size_t)BB*HHD * 2);
  float* gh       = (float*)alloc((size_t)BB*3*HHD * 4);
  float* hflow    = (float*)alloc((size_t)BB*CCH*TTOT * 4);
  bf16*  hflow_b  = (bf16*) alloc((size_t)BB*CCH*TTOT * 2);
  bf16*  ht_b     = (bf16*) alloc((size_t)BB*TTOT*CCH * 2);
  float* hct      = (float*)alloc((size_t)BB*TCV*CCH * 4);
  float* hc       = (float*)alloc((size_t)BB*CCH*TTOT * 4);
  bf16*  hc_b     = (bf16*) alloc((size_t)BB*CCH*TTOT * 2);
  float* sample   = (float*)alloc((size_t)BB*CCH*PPL * 4);
  float* muf      = (float*)alloc((size_t)BB*CCH*PPL * 4);
  float* sgf      = (float*)alloc((size_t)BB*CCH*PPL * 4);

  // ---- weight conversions ----
  k_cvt_bf16<<<CDIV((long)3*HHD*CCH,256),256,0,stream>>>(wih, wih_b, (long)3*HHD*CCH);
  k_cvt_bf16<<<CDIV((long)3*HHD*HHD,256),256,0,stream>>>(whh, whh_b, (long)3*HHD*HHD);
  k_cvt_bf16<<<CDIV((long)PPL*HHD,256),256,0,stream>>>(encmu_w, encmuw_b, (long)PPL*HHD);
  k_cvt_bf16<<<CDIV((long)PPL*HHD,256),256,0,stream>>>(encps_w, encpsw_b, (long)PPL*HHD);
  k_cvt_bf16<<<CDIV((long)NFL*PPL*TTOT,256),256,0,stream>>>(mu_w, muw_b, (long)NFL*PPL*TTOT);
  k_cvt_bf16<<<CDIV((long)NFL*PPL*TTOT,256),256,0,stream>>>(sg_w, sgw_b, (long)NFL*PPL*TTOT);
  k_pack_convw<<<CDIV((long)NFL*CCH*2*CCH,256),256,0,stream>>>(conv_w, convp_b);
  k_transpose_x<<<CDIV((long)BB*LLN*CCH,256),256,0,stream>>>(x_enc, xt_b);

  // ---- input projection: xw = x @ wih^T + bih  (10752 x 3072 x 512) ----
  {
    dim3 g(CDIV(3*HHD,BN), CDIV(BB*LLN,BM), 1);
    k_gemm_nt<<<g,256,0,stream>>>(xt_b, CCH, 0, wih_b, CCH, bih,
                                  xw, 3*HHD, 0, BB*LLN, 3*HHD, CCH);
  }

  // ---- GRU scan: 336 sequential (32 x 3072 x 1024) GEMM + gate steps ----
  k_zero_h<<<CDIV(BB*HHD,256),256,0,stream>>>(hgru, hgru_b);
  {
    dim3 g(CDIV(3*HHD,BN), CDIV(BB,BM), 1);
    for (int t = 0; t < LLN; ++t) {
      k_gemm_nt<<<g,256,0,stream>>>(hgru_b, HHD, 0, whh_b, HHD, bhh,
                                    gh, 3*HHD, 0, BB, 3*HHD, HHD);
      k_gru_gate<<<CDIV(BB*HHD,256),256,0,stream>>>(xw, gh, hgru, hgru_b, t);
    }
  }

  // ---- encoder head ----
  k_enc_build<<<CDIV((long)BB*CCH*HHD,256),256,0,stream>>>(hgru, wfx, bfx, hflow, hflow_b);
  {
    dim3 g(CDIV(PPL,BN), CDIV(BB*CCH,BM), 1);
    k_gemm_nt<<<g,256,0,stream>>>(hflow_b, TTOT, 0, encmuw_b, HHD, encmu_b,
                                  muf, PPL, 0, BB*CCH, PPL, HHD);
    k_gemm_nt<<<g,256,0,stream>>>(hflow_b, TTOT, 0, encpsw_b, HHD, encps_b,
                                  sgf, PPL, 0, BB*CCH, PPL, HHD);
  }
  k_sample_init<<<CDIV((long)BB*CCH*PPL,256),256,0,stream>>>(muf, sgf, eps, sample,
                                                             hflow, hflow_b);

  // ---- flow layers ----
  for (int l = 0; l < NFL; ++l) {
    k_t1<<<CDIV((long)BB*TTOT*CCH,256),256,0,stream>>>(hflow, ht_b);
    {
      // conv-as-GEMM: rowA = z*1216 + t (contiguous taps), rowC = z*1215 + t
      dim3 g(CDIV(CCH,BN), CDIV(TCV,BM), BB);
      k_gemm_nt<<<g,256,0,stream>>>(ht_b, CCH, TTOT,
                                    convp_b + (size_t)l*CCH*2*CCH, 2*CCH,
                                    conv_b + (size_t)l*CCH,
                                    hct, CCH, TCV, TCV, CCH, 2*CCH);
    }
    k_t2<<<CDIV((long)BB*CCH*TTOT,256),256,0,stream>>>(hct, hc, hc_b);
    {
      dim3 g(CDIV(PPL,BN), CDIV(BB*CCH,BM), 1);
      k_gemm_nt<<<g,256,0,stream>>>(hc_b, TTOT, 0,
                                    muw_b + (size_t)l*PPL*TTOT, TTOT,
                                    mu_b + (size_t)l*PPL,
                                    muf, PPL, 0, BB*CCH, PPL, TTOT);
      k_gemm_nt<<<g,256,0,stream>>>(hc_b, TTOT, 0,
                                    sgw_b + (size_t)l*PPL*TTOT, TTOT,
                                    sg_b + (size_t)l*PPL,
                                    sgf, PPL, 0, BB*CCH, PPL, TTOT);
    }
    k_flow_pw<<<CDIV((long)BB*CCH*TTOT,256),256,0,stream>>>(hc, muf, sgf, sample, hflow);
  }

  // ---- output transpose ----
  k_out_tr<<<CDIV((long)BB*PPL*CCH,256),256,0,stream>>>(sample, out);
}
